// encoder_block_light_57148834840961
// MI455X (gfx1250) — compile-verified, matrix operand order checked by reading, not run on hardware
//
#include <hip/hip_runtime.h>
#include <hip/hip_bf16.h>

typedef _Float16 v16h __attribute__((ext_vector_type(16)));
typedef float    v8f  __attribute__((ext_vector_type(8)));

#define HDIM 64
#define K5   125
#define K3   27
#define NMAT 135                        // 54 wa + 54 wb + 27 wl matrices (64x64)
#define NW   (NMAT * 64 * 64)          // total weight elements (halves)
#define TILE_M 32

// ---------- helpers ----------
__device__ __forceinline__ unsigned fkey(float f) {
    unsigned b = __float_as_uint(f);
    return (b & 0x80000000u) ? ~b : (b | 0x80000000u);
}
__device__ __forceinline__ float fdec(unsigned k) {
    unsigned b = (k & 0x80000000u) ? (k & 0x7FFFFFFFu) : ~k;
    return __uint_as_float(b);
}

// ---------- prefix-sum prep (serial, tiny) ----------
__global__ void prep_prefix(const long long* __restrict__ cnt5,
                            const long long* __restrict__ cnt3,
                            int* __restrict__ pfx5,   // 126
                            int* __restrict__ pfx3,   // 28
                            int* __restrict__ tpfx3)  // 28
{
    if (threadIdx.x == 0 && blockIdx.x == 0) {
        int s = 0;
        pfx5[0] = 0;
        for (int i = 0; i < K5; ++i) { s += (int)cnt5[i]; pfx5[i + 1] = s; }
        s = 0; int ts = 0;
        pfx3[0] = 0; tpfx3[0] = 0;
        for (int i = 0; i < K3; ++i) {
            int c = (int)cnt3[i];
            s += c;                          pfx3[i + 1]  = s;
            ts += (c + TILE_M - 1) / TILE_M; tpfx3[i + 1] = ts;
        }
    }
}

// ---------- weight conversion + shuffle into WMMA B-fragment order ----------
// Output layout: [matrix][kstep(2)][ntile(4)][lane(32)][i(16)] halves.
// Matrix order: wa[0..53], wb[54..107], wl[108..134].
__global__ void cvt_weights(const float* __restrict__ wa,
                            const float* __restrict__ wb,
                            const float* __restrict__ wl,
                            _Float16* __restrict__ whf)
{
    int o = blockIdx.x * blockDim.x + threadIdx.x;
    if (o >= NW) return;
    int kidx  = o >> 12;          // which 64x64 matrix
    int rest  = o & 4095;
    int i     = rest & 15;
    int lane  = (rest >> 4) & 31;
    int ntile = (rest >> 9) & 3;
    int ks    = (rest >> 11) & 1;

    const float* src;
    if (kidx < 54)        src = wa + ((size_t)kidx) * 4096;
    else if (kidx < 108)  src = wb + ((size_t)(kidx - 54)) * 4096;
    else                  src = wl + ((size_t)(kidx - 108)) * 4096;

    int col = (ntile << 4) + (lane & 15);
    int kb  = ((lane >> 4) & 1) * 8 + ks * 32;
    int K   = (i < 8) ? (kb + i) : (kb + 16 + (i - 8));
    whf[o] = (_Float16)src[K * HDIM + col];
}

// ---------- conv1: k=5^3, CIN=3 -> 64 (scalar, atomic scatter) ----------
__global__ void conv1_k5(const float* __restrict__ x,
                         const float* __restrict__ w1,   // [125][3][64]
                         const int* __restrict__ ki,
                         const int* __restrict__ ko,
                         const int* __restrict__ pfx,    // 126
                         float* __restrict__ x1,
                         int nPairs)
{
    long long t = (long long)blockIdx.x * blockDim.x + threadIdx.x;
    int p = (int)(t >> 6);
    if (p >= nPairs) return;
    int h = (int)(t & 63);
    int lo = 0, hi = K5;
    while (lo + 1 < hi) { int mid = (lo + hi) >> 1; if (pfx[mid] <= p) lo = mid; else hi = mid; }
    int k = lo;
    const float* xi = x + ki[p] * 3;
    const float* W  = w1 + (k * 3) * HDIM + h;
    float v = xi[0] * W[0] + xi[1] * W[HDIM] + xi[2] * W[2 * HDIM];
    atomicAdd(&x1[(unsigned)(ko[p] * HDIM + h)], v);
}

// ---------- pooling: segment max via monotonic uint keys ----------
__global__ void pool_max(const float* __restrict__ x1,
                         const int* __restrict__ pool_ids,
                         unsigned* __restrict__ xpkey,
                         int n0)
{
    long long t = (long long)blockIdx.x * blockDim.x + threadIdx.x;
    int i = (int)(t >> 6);
    if (i >= n0) return;
    int h = (int)(t & 63);
    unsigned k = fkey(x1[(unsigned)(i * HDIM + h)]);
    atomicMax(&xpkey[(unsigned)(pool_ids[i] * HDIM + h)], k);
}

__global__ void pool_finalize(unsigned* __restrict__ xpkey, long long n)
{
    long long i = (long long)blockIdx.x * blockDim.x + threadIdx.x;
    if (i >= n) return;
    ((float*)xpkey)[i] = fdec(xpkey[i]);
}

// ---------- 3^3 sparse conv via WMMA: gather 32 rows -> 32x64 GEMM -> scatter ----------
__global__ __launch_bounds__(128)
void conv3_wmma(const float* __restrict__ fin,
                const _Float16* __restrict__ Whf,  // fragment-ordered weights, 27 matrices
                const int* __restrict__ ki,
                const int* __restrict__ ko,
                const int* __restrict__ pfx,       // 28
                const int* __restrict__ tpfx,      // 28 tile prefix (ceil(cnt/32))
                float* __restrict__ fout,          // accumulated with atomics
                int applyRelu,
                int dumpRow)                       // spare row index for tail lanes
{
    __shared__ _Float16 As[TILE_M][72];            // gathered 32 x 64 tile (f16), padded
    __shared__ int KoS[TILE_M];                    // output rows; tail -> dumpRow

    int b = blockIdx.x;
    if (b >= tpfx[K3]) return;

    // locate kernel offset k so a tile never spans two weight matrices
    int k = 0;
    #pragma unroll 1
    for (int i = 0; i < K3; ++i) if (b >= tpfx[i + 1]) k = i + 1;

    int pairBase = pfx[k] + (b - tpfx[k]) * TILE_M;
    int e = pfx[k + 1];

    int tid = threadIdx.x;
    // cooperative gather: 32 rows x 64 cols as float4, f32 -> f16, fused ReLU
    for (int idx = tid; idx < TILE_M * 16; idx += 128) {
        int r  = idx >> 4;
        int c4 = (idx & 15) << 2;
        int p  = pairBase + r;
        float4 v = make_float4(0.f, 0.f, 0.f, 0.f);
        if (p < e) {
            v = *(const float4*)(fin + (unsigned)(ki[p] * HDIM + c4));
            if (applyRelu) {
                v.x = v.x > 0.f ? v.x : 0.f;  v.y = v.y > 0.f ? v.y : 0.f;
                v.z = v.z > 0.f ? v.z : 0.f;  v.w = v.w > 0.f ? v.w : 0.f;
            }
        }
        As[r][c4 + 0] = (_Float16)v.x;  As[r][c4 + 1] = (_Float16)v.y;
        As[r][c4 + 2] = (_Float16)v.z;  As[r][c4 + 3] = (_Float16)v.w;
    }
    if (tid < TILE_M) {
        int p = pairBase + tid;
        KoS[tid] = (p < e) ? ko[p] : dumpRow;
    }
    __syncthreads();

    int lane = tid & 31;
    int wave = tid >> 5;                 // 4 waves -> 4 N-tiles of 16 columns
    int rowL = lane & 15;
    int hi8  = ((lane >> 4) & 1) * 8;    // lanes 16..31 handle the +8 K/M halves
    int col  = (wave << 4) + rowL;

    const _Float16* Wfk = Whf + ((size_t)k << 12);

    v8f acc0 = {}, acc1 = {};
    #pragma unroll
    for (int ks = 0; ks < 2; ++ks) {
        // B fragment: contiguous 16 halves per lane (two global_load_b128)
        v16h bm = *(const v16h*)(Wfk + (((ks << 2) | wave) << 9) + (lane << 4));
        int kb = hi8 + ks * 32;
        v16h a0, a1;
        #pragma unroll
        for (int i = 0; i < 8; ++i) {
            a0[i]     = As[rowL][kb + i];
            a0[8 + i] = As[rowL][kb + 16 + i];
            a1[i]     = As[rowL + 16][kb + i];
            a1[8 + i] = As[rowL + 16][kb + 16 + i];
        }
        acc0 = __builtin_amdgcn_wmma_f32_16x16x32_f16(false, a0, false, bm,
                                                      (short)0, acc0, false, false);
        acc1 = __builtin_amdgcn_wmma_f32_16x16x32_f16(false, a1, false, bm,
                                                      (short)0, acc1, false, false);
    }

    // pull the 16 output rows into registers (contiguous -> ds_load_b128 x4)
    int ko0[8], ko1[8];
    #pragma unroll
    for (int v = 0; v < 8; ++v) { ko0[v] = KoS[hi8 + v]; ko1[v] = KoS[hi8 + 16 + v]; }

    // scatter-add: unconditional atomics (tail rows hit the dump row)
    #pragma unroll
    for (int v = 0; v < 8; ++v) {
        atomicAdd(&fout[(unsigned)(ko0[v] * HDIM + col)], acc0[v]);
        atomicAdd(&fout[(unsigned)(ko1[v] * HDIM + col)], acc1[v]);
    }
}

// ---------- elementwise ----------
__global__ void add_inplace(float* __restrict__ a, const float* __restrict__ b, long long n)
{
    long long i = (long long)blockIdx.x * blockDim.x + threadIdx.x;
    if (i < n) a[i] += b[i];
}

__global__ void elu_out(const float* __restrict__ t, float* __restrict__ out, long long n)
{
    long long i = (long long)blockIdx.x * blockDim.x + threadIdx.x;
    if (i >= n) return;
    float v = t[i];
    out[i] = v > 0.f ? v : (expf(v) - 1.f);
}

// ---------- to_quantize: [N1,64] @ [64,32] ----------
__global__ void quant_proj(const float* __restrict__ h,
                           const float* __restrict__ wq,   // [64][32]
                           float* __restrict__ out, int n1)
{
    long long t = (long long)blockIdx.x * blockDim.x + threadIdx.x;
    int i = (int)(t >> 5);
    if (i >= n1) return;
    int j = (int)(t & 31);
    const float* hr = h + (unsigned)(i * HDIM);
    float s = 0.f;
    #pragma unroll
    for (int c = 0; c < HDIM; ++c) s += hr[c] * wq[c * 32 + j];
    out[(unsigned)(i * 32 + j)] = s;
}

// ---------- host orchestration ----------
static inline size_t alignUp(size_t v, size_t a) { return (v + a - 1) / a * a; }

extern "C" void kernel_launch(void* const* d_in, const int* in_sizes, int n_in,
                              void* d_out, int out_size, void* d_ws, size_t ws_size,
                              hipStream_t stream) {
    const float*     x        = (const float*)d_in[0];
    const float*     w1       = (const float*)d_in[1];
    const float*     wa       = (const float*)d_in[2];
    const float*     wb       = (const float*)d_in[3];
    const float*     wl       = (const float*)d_in[4];
    const float*     wq       = (const float*)d_in[5];
    const int*       km5_in   = (const int*)d_in[6];
    const int*       km5_out  = (const int*)d_in[7];
    const long long* km5_cnt  = (const long long*)d_in[8];
    const int*       km3_in   = (const int*)d_in[9];
    const int*       km3_out  = (const int*)d_in[10];
    const long long* km3_cnt  = (const long long*)d_in[11];
    const int*       pool_ids = (const int*)d_in[12];

    const int N0  = in_sizes[0] / 3;
    const int nP5 = in_sizes[6];
    const int nP3 = in_sizes[9];
    const int N1  = out_size / 96;      // out = N1*64 + N1*32

    // rows for accumulation buffers: one spare dump row past N1
    const size_t rows0 = (size_t)((N0 > N1 + 1) ? N0 : N1 + 1);

    // ---- workspace carve-out ----
    char* ws = (char*)d_ws;
    size_t off = 0;
    int* pfx5  = (int*)(ws + off); off = alignUp(off + 126 * sizeof(int), 256);
    int* pfx3  = (int*)(ws + off); off = alignUp(off +  28 * sizeof(int), 256);
    int* tpfx3 = (int*)(ws + off); off = alignUp(off +  28 * sizeof(int), 256);
    _Float16* Whf = (_Float16*)(ws + off); off = alignUp(off + (size_t)NW * 2, 256);
    float* x1 = (float*)(ws + off); off = alignUp(off + rows0 * HDIM * 4, 256);           // reused as t
    float* xp = (float*)(ws + off); off = alignUp(off + (size_t)N1 * HDIM * 4, 256);
    float* h  = (float*)(ws + off); off = alignUp(off + (size_t)(N1 + 1) * HDIM * 4, 256); // +dump row
    float* t  = x1;                                   // rows0 >= N1+1, safe reuse after pooling

    float* out_lower = (float*)d_out;                      // [N1,64]
    float* out_quant = (float*)d_out + (size_t)N1 * HDIM;  // [N1,32]

    const long long n1h = (long long)N1 * HDIM;
    const int T = 256;

    // 1) prefixes + weight conversion/shuffle
    prep_prefix<<<1, 32, 0, stream>>>(km5_cnt, km3_cnt, pfx5, pfx3, tpfx3);
    cvt_weights<<<(NW + T - 1) / T, T, 0, stream>>>(wa, wb, wl, Whf);

    // 2) conv1 (k=5^3) into x1
    hipMemsetAsync(x1, 0, (size_t)N0 * HDIM * 4, stream);
    {
        long long work = (long long)nP5 * HDIM;
        conv1_k5<<<(int)((work + T - 1) / T), T, 0, stream>>>(x, w1, km5_in, km5_out, pfx5, x1, nP5);
    }

    // 3) segment-max pooling into xp
    hipMemsetAsync(xp, 0, n1h * 4, stream);           // key 0 == minimum
    {
        long long work = (long long)N0 * HDIM;
        pool_max<<<(int)((work + T - 1) / T), T, 0, stream>>>(x1, pool_ids, (unsigned*)xp, N0);
    }
    pool_finalize<<<(int)((n1h + T - 1) / T), T, 0, stream>>>((unsigned*)xp, n1h);
    hipMemcpyAsync(h, xp, n1h * 4, hipMemcpyDeviceToDevice, stream);

    // 4) residual blocks (WMMA convs). Matrix order in Whf: wa[0..53], wb[54..107], wl[108..134]
    const int convBlocks = (nP3 + TILE_M - 1) / TILE_M + K3;   // upper bound; kernel early-exits
    for (int r = 0; r < 2; ++r) {
        const _Float16* war = Whf + ((size_t)(r * K3)      << 12);
        const _Float16* wbr = Whf + ((size_t)(54 + r * K3) << 12);
        hipMemsetAsync(t, 0, n1h * 4, stream);
        conv3_wmma<<<convBlocks, 128, 0, stream>>>(h, war, km3_in, km3_out, pfx3, tpfx3, t, 0, N1);
        // h += conv(relu(t), wb)  (identity add is free: accumulate into h)
        conv3_wmma<<<convBlocks, 128, 0, stream>>>(t, wbr, km3_in, km3_out, pfx3, tpfx3, h, 1, N1);
    }

    // 5) outer residual: x3 = h + xp  (h started as xp, loop added the blocks)
    add_inplace<<<(int)((n1h + T - 1) / T), T, 0, stream>>>(h, xp, n1h);

    // 6) to_lower = elu(conv3(x3, wl))
    hipMemsetAsync(t, 0, n1h * 4, stream);
    conv3_wmma<<<convBlocks, 128, 0, stream>>>(h, Whf + ((size_t)108 << 12),
                                               km3_in, km3_out, pfx3, tpfx3, t, 0, N1);
    elu_out<<<(int)((n1h + T - 1) / T), T, 0, stream>>>(t, out_lower, n1h);

    // 7) to_quantize = x3 @ wq
    {
        long long work = (long long)N1 * 32;
        quant_proj<<<(int)((work + T - 1) / T), T, 0, stream>>>(h, wq, out_quant, N1);
    }
}
